// bnn_54631984005843
// MI455X (gfx1250) — matrix operand order repaired and checked by reading, not compiled, not run on hardware
//
#include <hip/hip_runtime.h>

typedef __attribute__((ext_vector_type(16))) _Float16     v16h;
typedef __attribute__((ext_vector_type(8)))  float        v8f;
typedef __attribute__((ext_vector_type(8)))  unsigned int v8u;
typedef __attribute__((ext_vector_type(4)))  unsigned int v4u;
typedef __attribute__((ext_vector_type(2)))  _Float16     h2;

template<bool B> struct BoolC { static constexpr bool value = B; };

#define MC_NS   150
#define BATCH   64
#define HID     512
#define D0      784
#define D0P     800                 // K padded to 25 tiles of 32
#define XSTR    808                 // x LDS row stride: 404 dw, %64=20 -> no bank conflicts
#define HSTR    520                 // h LDS row stride: 260 dw, %64=4  -> no bank conflicts
#define LDS_A   (BATCH * XSTR * 2)  // 103424 B (x, reused as h buffer)
#define LDS_B   (BATCH * HSTR * 2)  // 66560 B
#define LDS_TOT (LDS_A + LDS_B)     // 169984 B < 320 KB WGP LDS

// f16-element counts / workspace offsets (halfs) for precomputed mu/sigma
#define NW0 (HID * D0)
#define NW1 (HID * HID)
#define NW2 (HID * HID)
#define NW3 (10 * HID)
#define OFF_MU0 0
#define OFF_SG0 (OFF_MU0 + NW0)
#define OFF_MU1 (OFF_SG0 + NW0)
#define OFF_SG1 (OFF_MU1 + NW1)
#define OFF_MU2 (OFF_SG1 + NW1)
#define OFF_SG2 (OFF_MU2 + NW2)
#define OFF_MU3 (OFF_SG2 + NW2)
#define OFF_SG3 (OFF_MU3 + NW3)

// ---------------- exact-ish gauss for the (tiny) bias path ------------------
__device__ __forceinline__ unsigned int bnn_hash(unsigned int v) {
    v ^= v >> 17; v *= 0xed5ad4bbu;
    v ^= v >> 11; v *= 0xac4c1b51u;
    v ^= v >> 15; v *= 0x31848babu;
    v ^= v >> 14;
    return v;
}
__device__ __forceinline__ float bnn_gauss(unsigned int key) {
    unsigned int h1 = bnn_hash(key);
    unsigned int h2v = bnn_hash(key ^ 0x9e3779b9u);
    float u1 = ((float)(h1 >> 8) + 1.0f) * (1.0f / 16777216.0f);
    float u2 = (float)(h2v >> 8) * (1.0f / 16777216.0f);
    return __fsqrt_rn(-2.0f * __logf(u1)) * __cosf(6.28318530718f * u2);
}

// ---------------- pre-pass: mu,v (f32) -> mu,sigma=exp(v) (f16) -------------
__global__ void __launch_bounds__(256)
bnn_prep_kernel(const float* __restrict__ mu, const float* __restrict__ v,
                _Float16* __restrict__ muh, _Float16* __restrict__ sgh, int n) {
    int i = blockIdx.x * 256 + threadIdx.x;
    if (i < n) {
        muh[i] = (_Float16)mu[i];
        sgh[i] = (_Float16)__expf(v[i]);
    }
}

// ---------------- fused sampled-GEMM layer ----------------------------------
// out = act(in[64xKDIM] * W^T + b), W = mu + sigma*eps, eps drawn per
// (sample, element) by a counter-based packed-f16 triangular sampler.
template<int KDIM, int KP, int SIN, int NTILES, bool RELU, bool TO_GLOBAL>
__device__ __forceinline__ void bnn_layer(
    const _Float16* __restrict__ inLds,   // [64 x SIN] f16
    _Float16*       __restrict__ outLds,  // [64 x HSTR] f16 (if !TO_GLOBAL)
    float*          __restrict__ gout,    // [NS x 64 x NOUT] f32 (if TO_GLOBAL)
    const _Float16* __restrict__ muh, const _Float16* __restrict__ sgh,
    const float*    __restrict__ muB, const float* __restrict__ vB,
    int sample, unsigned int layer, int NOUT)
{
    const int tid  = threadIdx.x;
    const int wave = tid >> 5;
    const int lane = tid & 31;
    const int nlo  = lane & 15;          // B col / A row / C col per lane
    const int hi   = lane >> 4;
    const int khB  = hi * 16;            // B-frag K offset for this lane
    const int khA  = hi * 8;             // unused; A-frag also loads at khB (k = 16*hi + s)
    constexpr int KT     = KP / 32;      // total K-tiles
    constexpr int KTFULL = KDIM / 32;    // fully in-range K-tiles (24 or 16)

    const unsigned int sseed = (unsigned int)sample * 0x01000193u ^ (layer << 27);

    for (int nt = wave; nt < NTILES; nt += 8) {
        const int o  = nt * 16 + nlo;
        // clamp addressing for the 10-row output layer (lanes 10..15 discarded)
        const int oc = TO_GLOBAL ? (o < NOUT ? o : NOUT - 1) : o;
        const _Float16* __restrict__ muRow = muh + (size_t)oc * KDIM;
        const _Float16* __restrict__ sgRow = sgh + (size_t)oc * KDIM;

        v8f acc[4];
        #pragma unroll
        for (int mt = 0; mt < 4; ++mt) acc[mt] = (v8f){0.f,0.f,0.f,0.f,0.f,0.f,0.f,0.f};

        // one K-tile: sample B fragment, load 4 A fragments, 4 WMMAs
        auto ktile = [&](int kt, auto guard_c) {
            constexpr bool GUARD = decltype(guard_c)::value;
            const int kb = kt * 32 + khB;
            if (kt * 32 + 96 <= KDIM) {              // uniform-branch prefetch
                __builtin_prefetch(muRow + kt * 32 + 64, 0, 0);
                __builtin_prefetch(sgRow + kt * 32 + 64, 0, 0);
            }
            v8u bw;
            if (GUARD) bw = (v8u){0u,0u,0u,0u,0u,0u,0u,0u};
            // tail guard is all-or-nothing per lane (kb, KDIM multiples of 16)
            if (!GUARD || kb + 16 <= KDIM) {
                const v4u* __restrict__ mp =
                    (const v4u*)__builtin_assume_aligned(muRow + kb, 16);
                const v4u* __restrict__ sp =
                    (const v4u*)__builtin_assume_aligned(sgRow + kb, 16);
                const v4u m0 = mp[0], m1 = mp[1];    // global_load_b128 x2
                const v4u s0 = sp[0], s1 = sp[1];    // global_load_b128 x2
                const unsigned int kbase = sseed + (unsigned int)(oc * KDIM + kb);
                #pragma unroll
                for (int p = 0; p < 8; ++p) {
                    unsigned int hA = (kbase + 2u * p) * 0x9E3779B1u;
                    hA ^= hA >> 16; hA *= 0x85EBCA77u; hA ^= hA >> 13;
                    unsigned int hB = hA * 0xC2B2AE3Du; hB ^= hB >> 16;
                    // two f16 uniforms in [1,2) per hash via mantissa bits
                    h2 u1 = __builtin_bit_cast(h2, (hA & 0x03FF03FFu) | 0x3C003C00u);
                    h2 u2 = __builtin_bit_cast(h2, (hB & 0x03FF03FFu) | 0x3C003C00u);
                    // triangular eps, unit variance: (u1+u2-3)*sqrt(6)
                    h2 eps = (u1 + u2) * (_Float16)2.44949f - (_Float16)7.34847f;
                    const unsigned int mu2u = (p < 4) ? m0[p] : m1[p - 4];
                    const unsigned int sg2u = (p < 4) ? s0[p] : s1[p - 4];
                    h2 w2 = __builtin_bit_cast(h2, sg2u) * eps
                          + __builtin_bit_cast(h2, mu2u);     // v_pk_fma_f16
                    bw[p] = __builtin_bit_cast(unsigned int, w2);
                }
            }
            const v16h bf = __builtin_bit_cast(v16h, bw);

            v16h afr[4];
            #pragma unroll
            for (int mt = 0; mt < 4; ++mt) {
                const int row = mt * 16 + nlo;
                const _Float16* __restrict__ arow = inLds + (size_t)row * SIN + kt * 32;
                #pragma unroll
                for (int j = 0; j < 4; ++j) {
                    afr[mt][2*j]     = arow[khB + 2*j];
                    afr[mt][2*j + 1] = arow[khB + 2*j + 1];
                }
                #pragma unroll
                for (int j = 0; j < 4; ++j) {
                    afr[mt][8 + 2*j] = arow[khB + 8 + 2*j];
                    afr[mt][9 + 2*j] = arow[khB + 8 + 2*j + 1];
                }
            }
            #pragma unroll
            for (int mt = 0; mt < 4; ++mt) {
                acc[mt] = __builtin_amdgcn_wmma_f32_16x16x32_f16(
                    false, afr[mt], false, bf, (short)0, acc[mt], false, false);
            }
        };

        for (int kt = 0; kt < KTFULL; ++kt)          // hot loop: no exec masking
            ktile(kt, BoolC<false>{});
        if constexpr (KT > KTFULL)                   // layer-0 half tile only
            ktile(KTFULL, BoolC<true>{});

        // ---- sampled bias (tiny; exact Box-Muller), epilogue, writeback
        const unsigned int bkey = (0x40000000u + (unsigned int)oc) * 2654435761u
                                ^ sseed ^ 0x5bd1e995u;
        const float bias = __builtin_fmaf(__expf(vB[oc]), bnn_gauss(bkey), muB[oc]);

        #pragma unroll
        for (int mt = 0; mt < 4; ++mt) {
            #pragma unroll
            for (int r = 0; r < 8; ++r) {
                const int row = mt * 16 + r + (hi << 3);
                float val = acc[mt][r] + bias;
                if (RELU) val = fmaxf(val, 0.0f);
                if (TO_GLOBAL) {
                    if (nlo < NOUT)
                        gout[((size_t)sample * BATCH + row) * NOUT + nlo] = val;
                } else {
                    outLds[(size_t)row * HSTR + o] = (_Float16)val;
                }
            }
        }
    }
}

// ---------------- full 4-layer MC forward: one workgroup per MC sample ------
__global__ void __launch_bounds__(256, 1)
bnn_mc_forward_kernel(
    const float* __restrict__ x,
    const _Float16* __restrict__ muW0h, const _Float16* __restrict__ sgW0h,
    const _Float16* __restrict__ muW1h, const _Float16* __restrict__ sgW1h,
    const _Float16* __restrict__ muW2h, const _Float16* __restrict__ sgW2h,
    const _Float16* __restrict__ muW3h, const _Float16* __restrict__ sgW3h,
    const float* __restrict__ muB0, const float* __restrict__ vB0,
    const float* __restrict__ muB1, const float* __restrict__ vB1,
    const float* __restrict__ muB2, const float* __restrict__ vB2,
    const float* __restrict__ muB3, const float* __restrict__ vB3,
    float* __restrict__ out)
{
    extern __shared__ char smem[];
    _Float16* bufA = (_Float16*)smem;            // x (64x808), later h (64x520)
    _Float16* bufB = (_Float16*)(smem + LDS_A);  // h (64x520)

    const int s   = blockIdx.x;
    const int tid = threadIdx.x;

    for (int i = tid; i < BATCH * XSTR; i += 256) {  // x -> LDS f16, pad 784->808
        const int r = i / XSTR, c = i - r * XSTR;
        bufA[i] = (_Float16)((c < D0) ? x[r * D0 + c] : 0.0f);
    }
    __syncthreads();

    bnn_layer<D0,  D0P, XSTR, 32, true,  false>(bufA, bufB, nullptr, muW0h, sgW0h, muB0, vB0, s, 0u, HID);
    __syncthreads();
    bnn_layer<HID, HID, HSTR, 32, true,  false>(bufB, bufA, nullptr, muW1h, sgW1h, muB1, vB1, s, 1u, HID);
    __syncthreads();
    bnn_layer<HID, HID, HSTR, 32, true,  false>(bufA, bufB, nullptr, muW2h, sgW2h, muB2, vB2, s, 2u, HID);
    __syncthreads();
    bnn_layer<HID, HID, HSTR, 1,  false, true >(bufB, nullptr, out,  muW3h, sgW3h, muB3, vB3, s, 3u, 10);
}

extern "C" void kernel_launch(void* const* d_in, const int* in_sizes, int n_in,
                              void* d_out, int out_size, void* d_ws, size_t ws_size,
                              hipStream_t stream) {
    (void)in_sizes; (void)n_in; (void)out_size; (void)ws_size;
    const float* x    = (const float*)d_in[0];
    const float* muW0 = (const float*)d_in[1];  const float* muB0 = (const float*)d_in[2];
    const float* muW1 = (const float*)d_in[3];  const float* muB1 = (const float*)d_in[4];
    const float* muW2 = (const float*)d_in[5];  const float* muB2 = (const float*)d_in[6];
    const float* muW3 = (const float*)d_in[7];  const float* muB3 = (const float*)d_in[8];
    const float* vW0  = (const float*)d_in[9];  const float* vB0  = (const float*)d_in[10];
    const float* vW1  = (const float*)d_in[11]; const float* vB1  = (const float*)d_in[12];
    const float* vW2  = (const float*)d_in[13]; const float* vB2  = (const float*)d_in[14];
    const float* vW3  = (const float*)d_in[15]; const float* vB3  = (const float*)d_in[16];
    float* out = (float*)d_out;

    _Float16* ws = (_Float16*)d_ws;
    _Float16 *muW0h = ws + OFF_MU0, *sgW0h = ws + OFF_SG0;
    _Float16 *muW1h = ws + OFF_MU1, *sgW1h = ws + OFF_SG1;
    _Float16 *muW2h = ws + OFF_MU2, *sgW2h = ws + OFF_SG2;
    _Float16 *muW3h = ws + OFF_MU3, *sgW3h = ws + OFF_SG3;

    // pre-pass: fold exp(v) once (sample-invariant), downconvert to f16 in L2
    bnn_prep_kernel<<<(NW0 + 255) / 256, 256, 0, stream>>>(muW0, vW0, muW0h, sgW0h, NW0);
    bnn_prep_kernel<<<(NW1 + 255) / 256, 256, 0, stream>>>(muW1, vW1, muW1h, sgW1h, NW1);
    bnn_prep_kernel<<<(NW2 + 255) / 256, 256, 0, stream>>>(muW2, vW2, muW2h, sgW2h, NW2);
    bnn_prep_kernel<<<(NW3 + 255) / 256, 256, 0, stream>>>(muW3, vW3, muW3h, sgW3h, NW3);

    static bool attr_set = false;   // host-side launch config only
    if (!attr_set) {
        hipFuncSetAttribute((const void*)bnn_mc_forward_kernel,
                            hipFuncAttributeMaxDynamicSharedMemorySize, LDS_TOT);
        attr_set = true;
    }
    bnn_mc_forward_kernel<<<dim3(MC_NS), dim3(256), LDS_TOT, stream>>>(
        x, muW0h, sgW0h, muW1h, sgW1h, muW2h, sgW2h, muW3h, sgW3h,
        muB0, vB0, muB1, vB1, muB2, vB2, muB3, vB3, out);
}
